// MultiHeadAttention_57569741636379
// MI455X (gfx1250) — matrix operand order repaired and checked by reading, not compiled, not run on hardware
//
#include <hip/hip_runtime.h>
#include <hip/hip_bf16.h>

// ---------------- problem constants ----------------
constexpr int Bb = 2, Ss = 2048, Ee = 1024, Hh = 16, Dh = 64;
constexpr int BS = Bb * Ss;            // 4096 rows
constexpr long MiB = 1024 * 1024;

// ---------------- vector / fragment types ----------------
typedef __attribute__((ext_vector_type(16))) __bf16         v16bf;
typedef __attribute__((ext_vector_type(8)))  float          v8f;
typedef __attribute__((ext_vector_type(4)))  unsigned int   v4u;
typedef __attribute__((ext_vector_type(4)))  float          v4f;
typedef __attribute__((ext_vector_type(4)))  unsigned short v4s;

// A/B fragment = two contiguous 16-byte runs (ISA 7.12.2 layouts)
union frag_u { v4u q[2]; v16bf b; };

static __device__ __forceinline__ unsigned short f2bf(float f) {
  unsigned int x = __float_as_uint(f);
  x += 0x7FFFu + ((x >> 16) & 1u);          // round-to-nearest-even
  return (unsigned short)(x >> 16);
}

// CDNA5 async copy: global -> LDS, 16B per lane, tracked by ASYNCcnt.
static __device__ __forceinline__ void async_g2l_b128(void* lds, const void* gaddr) {
  unsigned       l = (unsigned)(unsigned long)lds;   // LDS aperture: low 32b = offset
  unsigned long  g = (unsigned long)gaddr;
  asm volatile("global_load_async_to_lds_b128 %0, %1, off"
               :: "v"(l), "v"(g) : "memory");
}
static __device__ __forceinline__ void wait_asynccnt0() {
  asm volatile("s_wait_asynccnt 0x0" ::: "memory");
}

// ---------------- 1) f32 -> bf16, 4 elements/thread ----------------
__global__ void mha_cvt_bf16(const v4f* __restrict__ src,
                             v4s* __restrict__ dst, int n4) {
  int i = blockIdx.x * blockDim.x + threadIdx.x;
  if (i < n4) {
    v4f x = src[i];
    v4s o;
    #pragma unroll
    for (int k = 0; k < 4; ++k) o[k] = f2bf(x[k]);
    dst[i] = o;
  }
}

// ---------------- 2) W[K,N] f32 -> Wt[N,K] bf16 (tiled transpose) --------
__global__ __launch_bounds__(256) void mha_transpose_cvt(
    const float* __restrict__ W, unsigned short* __restrict__ Wt) {
  __shared__ float tile[32][33];
  int bx = blockIdx.x & 31;       // N tile
  int by = blockIdx.x >> 5;       // K tile
  int tx = threadIdx.x & 31;
  int ty = threadIdx.x >> 5;      // 0..7
  #pragma unroll
  for (int i = 0; i < 4; ++i)
    tile[ty + i * 8][tx] = W[(by * 32 + ty + i * 8) * Ee + bx * 32 + tx];
  __syncthreads();
  #pragma unroll
  for (int i = 0; i < 4; ++i)
    Wt[(bx * 32 + ty + i * 8) * Ee + by * 32 + tx] = f2bf(tile[tx][ty + i * 8]);
}

// ---------------- 3) WMMA GEMM, 64x64 per wave (4x4 register blocking) ----
// Y = scale * (A[M,K]bf16 @ Bt[N,K]bf16^T + bias)
// MODE 0: bf16 head-split [B,H,S,Dh] ; MODE 1: bf16 transposed [B,H,Dh,S] ;
// MODE 2: f32 row-major [M,N]
template <int MODE>
__global__ __launch_bounds__(128) void mha_gemm_bf16(
    const unsigned short* __restrict__ A, const unsigned short* __restrict__ Bt,
    const float* __restrict__ bias, void* __restrict__ outp, float scale) {
  const int lane = threadIdx.x & 31;
  const int wave = threadIdx.x >> 5;
  const int half = lane >> 4;
  const int l15  = lane & 15;
  const int tile = blockIdx.x * 4 + wave;     // 1024 tiles = 64 (M) x 16 (N)
  const int tm = tile >> 4;
  const int tn = tile & 15;

  const v4u* Aq = (const v4u*)A;      // 16-byte units, row = Ee/8 units
  const v4u* Bq = (const v4u*)Bt;

  v8f acc[4][4];
  #pragma unroll
  for (int i = 0; i < 4; ++i)
    #pragma unroll
    for (int j = 0; j < 4; ++j)
      acc[i][j] = (v8f){0.f,0.f,0.f,0.f,0.f,0.f,0.f,0.f};

  int aRow[4], bRow[4];               // per-lane row bases in 16B units
  #pragma unroll
  for (int i = 0; i < 4; ++i) aRow[i] = (tm * 64 + i * 16 + l15) * (Ee >> 3);
  #pragma unroll
  for (int j = 0; j < 4; ++j) bRow[j] = (tn * 64 + j * 16 + l15) * (Ee >> 3);

  #pragma unroll 1
  for (int kc8 = 0; kc8 < (Ee >> 3); kc8 += 4) {   // K step of 32 elements
    if (kc8 + 4 < (Ee >> 3)) {                     // CDNA5 global_prefetch_b8
      __builtin_prefetch((const char*)(Aq + aRow[0] + kc8 + 4), 0, 1);
      __builtin_prefetch((const char*)(Bq + bRow[0] + kc8 + 4), 0, 1);
    }
    frag_u af[4], bf[4];
    #pragma unroll
    for (int i = 0; i < 4; ++i) {                  // A: runs at k=half*8, 16+half*8
      af[i].q[0] = Aq[aRow[i] + kc8 + half];
      af[i].q[1] = Aq[aRow[i] + kc8 + half + 2];
    }
    #pragma unroll
    for (int j = 0; j < 4; ++j) {                  // B: runs at k=half*8, 16+half*8
      bf[j].q[0] = Bq[bRow[j] + kc8 + half];
      bf[j].q[1] = Bq[bRow[j] + kc8 + half + 2];
    }
    #pragma unroll
    for (int i = 0; i < 4; ++i)
      #pragma unroll
      for (int j = 0; j < 4; ++j)
        acc[i][j] = __builtin_amdgcn_wmma_f32_16x16x32_bf16(
            false, af[i].b, false, bf[j].b, (short)0, acc[i][j], false, false);
  }

  float bb[4];
  #pragma unroll
  for (int j = 0; j < 4; ++j) bb[j] = bias[tn * 64 + j * 16 + l15];

  if (MODE == 2) {
    float* O = (float*)outp;
    #pragma unroll
    for (int i = 0; i < 4; ++i) {
      const int mbase = tm * 64 + i * 16 + 8 * half;
      #pragma unroll
      for (int j = 0; j < 4; ++j) {
        const int n = tn * 64 + j * 16 + l15;
        #pragma unroll
        for (int r = 0; r < 8; ++r)
          O[(long)(mbase + r) * Ee + n] = (acc[i][j][r] + bb[j]) * scale;
      }
    }
  } else {
    // each wave's 64 columns are exactly head h == tn
    unsigned short* O = (unsigned short*)outp;
    #pragma unroll
    for (int i = 0; i < 4; ++i) {
      const int base = tm * 64 + i * 16;
      const int bh = (base >> 11) * Hh + tn;
      const int sb = (base & (Ss - 1)) + 8 * half;
      #pragma unroll
      for (int j = 0; j < 4; ++j) {
        const int d = j * 16 + l15;
        #pragma unroll
        for (int r = 0; r < 8; ++r) {
          long idx = (MODE == 0) ? ((long)(bh * Ss + sb + r) * Dh + d)
                                 : ((long)(bh * Dh + d) * Ss + sb + r);
          O[idx] = f2bf((acc[i][j][r] + bb[j]) * scale);
        }
      }
    }
  }
}

// ---------------- 4) flash attention per 16-row query tile ----------------
// Qh (pre-scaled by 1/sqrt(Dh)), Kh: [B,H,S,Dh] bf16 ; Vt: [B,H,Dh,S] bf16
// Ctx: [B,S,E] bf16. All 4 waves of a block share (b,h): K/V chunks staged
// once into LDS via CDNA5 async global->LDS copies (ASYNCcnt).
__global__ __launch_bounds__(128) void mha_attn(
    const unsigned short* __restrict__ Qh, const unsigned short* __restrict__ Kh,
    const unsigned short* __restrict__ Vt, const int* __restrict__ mask,
    unsigned short* __restrict__ Ctx) {
  __shared__ __align__(16) unsigned short Ksh[32 * 64];   // [key][d]  4 KB
  __shared__ __align__(16) unsigned short Vsh[64 * 32];   // [d][key]  4 KB
  __shared__ __align__(16) unsigned short Pbuf[4][512];   // per-wave 16x32 P
  const int tid  = threadIdx.x;
  const int lane = tid & 31;
  const int wave = tid >> 5;
  const int half = lane >> 4;
  const int l15  = lane & 15;
  const int wt = blockIdx.x * 4 + wave;             // 4096 = B*H * (S/16)
  const int qt = wt & 127;
  const int bh = wt >> 7;                           // identical across the block
  const int b  = bh >> 4;
  const int h  = bh & 15;

  const v4u* Qq = (const v4u*)Qh;                   // rows = Dh/8 = 8 units
  const v4u* Kq = (const v4u*)Kh;
  const v4u* Vq = (const v4u*)Vt;                   // rows = Ss/8 = 256 units
  v4u* KshQ = (v4u*)Ksh;
  v4u* VshQ = (v4u*)Vsh;
  unsigned short* Ps = Pbuf[wave];
  const v4u* Pq = (const v4u*)Pbuf[wave];           // P row = 4 units

  // Q fragments for d-chunks 0..31, 32..63 (reused across all key chunks)
  frag_u qf[2];
  const int qrow = (bh * Ss + qt * 16 + l15) * (Dh >> 3);
  #pragma unroll
  for (int c = 0; c < 2; ++c) {
    qf[c].q[0] = Qq[qrow + c * 4 + half];
    qf[c].q[1] = Qq[qrow + c * 4 + half + 2];
  }

  v8f ctx0 = {0,0,0,0,0,0,0,0}, ctx1 = ctx0, ctx2 = ctx0, ctx3 = ctx0;
  float mrow[8], lrow[8];
  #pragma unroll
  for (int r = 0; r < 8; ++r) { mrow[r] = -1e30f; lrow[r] = 0.f; }

  // staging indices for this thread (constant across chunks)
  const int ku0 = tid,        ku1 = tid + 128;      // K: 256 16B units
  const int kr0 = ku0 >> 3,   ko0 = ku0 & 7;
  const int kr1 = ku1 >> 3,   ko1 = ku1 & 7;
  const int vr0 = ku0 >> 2,   vo0 = ku0 & 3;        // V: 256 16B units
  const int vr1 = ku1 >> 2,   vo1 = ku1 & 3;

  #pragma unroll 1
  for (int kc = 0; kc < Ss; kc += 32) {
    // ---- async staging of K (32x64) and V (64x32) chunks into LDS
    async_g2l_b128(&KshQ[ku0], &Kq[(bh * Ss + kc + kr0) * (Dh >> 3) + ko0]);
    async_g2l_b128(&KshQ[ku1], &Kq[(bh * Ss + kc + kr1) * (Dh >> 3) + ko1]);
    async_g2l_b128(&VshQ[ku0], &Vq[(bh * Dh + vr0) * (Ss >> 3) + (kc >> 3) + vo0]);
    async_g2l_b128(&VshQ[ku1], &Vq[(bh * Dh + vr1) * (Ss >> 3) + (kc >> 3) + vo1]);
    if (kc + 32 < Ss) {                             // prefetch next chunk to L2
      __builtin_prefetch((const char*)(Kq + (bh * Ss + kc + 32 + (tid >> 3)) * (Dh >> 3)), 0, 1);
      __builtin_prefetch((const char*)(Vq + (bh * Dh + (tid >> 1)) * (Ss >> 3) + ((kc + 32) >> 3)), 0, 1);
    }
    wait_asynccnt0();                               // our async copies landed
    __syncthreads();                                // whole block's copies landed

    // ---- scores: two 16-wide key tiles, contracted over Dh=64 (2 WMMAs each)
    v8f sc[2];
    #pragma unroll
    for (int j = 0; j < 2; ++j) {
      v8f s = {0,0,0,0,0,0,0,0};
      const int krl = (j * 16 + l15) * 8;           // local key row, 16B units
      #pragma unroll
      for (int c = 0; c < 2; ++c) {
        frag_u kf;
        kf.q[0] = KshQ[krl + c * 4 + half];
        kf.q[1] = KshQ[krl + c * 4 + half + 2];
        s = __builtin_amdgcn_wmma_f32_16x16x32_bf16(
            false, qf[c].b, false, kf.b, (short)0, s, false, false);
      }
      sc[j] = s;
    }
    // ---- mask (scores already scaled via pre-scaled Q)
    const int mk0 = mask[b * Ss + kc + l15];
    const int mk1 = mask[b * Ss + kc + 16 + l15];
    #pragma unroll
    for (int r = 0; r < 8; ++r) {
      sc[0][r] = mk0 ? sc[0][r] : -1e9f;
      sc[1][r] = mk1 ? sc[1][r] : -1e9f;
    }
    // ---- online softmax per query row (row = r + 8*half, across 16 lanes)
    #pragma unroll
    for (int r = 0; r < 8; ++r) {
      float tmax = fmaxf(sc[0][r], sc[1][r]);
      #pragma unroll
      for (int mm = 1; mm < 16; mm <<= 1)
        tmax = fmaxf(tmax, __shfl_xor(tmax, mm, 16));
      float mnew = fmaxf(mrow[r], tmax);
      float corr = __expf(mrow[r] - mnew);
      mrow[r] = mnew;
      float p0 = __expf(sc[0][r] - mnew);
      float p1 = __expf(sc[1][r] - mnew);
      float rs = p0 + p1;
      #pragma unroll
      for (int mm = 1; mm < 16; mm <<= 1)
        rs += __shfl_xor(rs, mm, 16);
      lrow[r] = lrow[r] * corr + rs;
      ctx0[r] *= corr; ctx1[r] *= corr; ctx2[r] *= corr; ctx3[r] *= corr;
      int prow = r + 8 * half;
      Ps[prow * 32 + l15]      = f2bf(p0);
      Ps[prow * 32 + 16 + l15] = f2bf(p1);
    }
    // ---- reload P as bf16 A-fragment (intra-wave LDS ops are in-order)
    frag_u pf;
    pf.q[0] = Pq[l15 * 4 + half];        // k = half*8 .. +7
    pf.q[1] = Pq[l15 * 4 + half + 2];    // k = 16+half*8 .. +7
    // ---- ctx += P(16x32) @ V(32x64): 4 column tiles
    #pragma unroll
    for (int t = 0; t < 4; ++t) {
      frag_u vf;
      const int vrl = (t * 16 + l15) * 4;           // local d row, 16B units
      vf.q[0] = VshQ[vrl + half];
      vf.q[1] = VshQ[vrl + half + 2];
      v8f* cp = (t == 0) ? &ctx0 : (t == 1) ? &ctx1 : (t == 2) ? &ctx2 : &ctx3;
      *cp = __builtin_amdgcn_wmma_f32_16x16x32_bf16(
          false, pf.b, false, vf.b, (short)0, *cp, false, false);
    }
    __syncthreads();                                // protect Ksh/Vsh (WAR)
  }

  // ---- normalize and write merged-head context [B,S,E] bf16
  #pragma unroll
  for (int r = 0; r < 8; ++r) {
    float inv = 1.0f / lrow[r];
    int srow = qt * 16 + r + 8 * half;
    long base = (long)(b * Ss + srow) * Ee + h * Dh + l15;
    Ctx[base +  0] = f2bf(ctx0[r] * inv);
    Ctx[base + 16] = f2bf(ctx1[r] * inv);
    Ctx[base + 32] = f2bf(ctx2[r] * inv);
    Ctx[base + 48] = f2bf(ctx3[r] * inv);
  }
}

// ---------------- host-side launch ----------------
extern "C" void kernel_launch(void* const* d_in, const int* in_sizes, int n_in,
                              void* d_out, int out_size, void* d_ws, size_t ws_size,
                              hipStream_t stream) {
  const float* query = (const float*)d_in[0];
  const float* key   = (const float*)d_in[1];
  const float* value = (const float*)d_in[2];
  const int*   maskp = (const int*)d_in[3];
  const float* Wq = (const float*)d_in[4];  const float* bq = (const float*)d_in[5];
  const float* Wk = (const float*)d_in[6];  const float* bk = (const float*)d_in[7];
  const float* Wv = (const float*)d_in[8];  const float* bv = (const float*)d_in[9];
  const float* Wo = (const float*)d_in[10]; const float* bo = (const float*)d_in[11];

  unsigned char* w = (unsigned char*)d_ws;
  unsigned short* Xq  = (unsigned short*)(w + 0 * MiB);   // 8 MiB  [BS,E] bf16
  unsigned short* Xk  = (unsigned short*)(w + 8 * MiB);
  unsigned short* Xv  = (unsigned short*)(w + 16 * MiB);
  unsigned short* Wqb = (unsigned short*)(w + 24 * MiB);  // 2 MiB each [N,K] bf16
  unsigned short* Wkb = (unsigned short*)(w + 26 * MiB);
  unsigned short* Wvb = (unsigned short*)(w + 28 * MiB);
  unsigned short* Wob = (unsigned short*)(w + 30 * MiB);
  unsigned short* Qh  = (unsigned short*)(w + 32 * MiB);  // 8 MiB [B,H,S,Dh]
  unsigned short* Kh  = (unsigned short*)(w + 40 * MiB);
  unsigned short* Vt  = (unsigned short*)(w + 48 * MiB);  // 8 MiB [B,H,Dh,S]
  unsigned short* Ctx = Xq;  // Xq dead after the Q projection -> reuse

  const int nElem = BS * Ee;                    // 4,194,304
  const int n4 = nElem / 4;
  dim3 cvtGrid(n4 / 256);
  mha_cvt_bf16<<<cvtGrid, 256, 0, stream>>>((const v4f*)query, (v4s*)Xq, n4);
  mha_cvt_bf16<<<cvtGrid, 256, 0, stream>>>((const v4f*)key,   (v4s*)Xk, n4);
  mha_cvt_bf16<<<cvtGrid, 256, 0, stream>>>((const v4f*)value, (v4s*)Xv, n4);

  mha_transpose_cvt<<<1024, 256, 0, stream>>>(Wq, Wqb);
  mha_transpose_cvt<<<1024, 256, 0, stream>>>(Wk, Wkb);
  mha_transpose_cvt<<<1024, 256, 0, stream>>>(Wv, Wvb);
  mha_transpose_cvt<<<1024, 256, 0, stream>>>(Wo, Wob);

  // 1024 wave tiles (64x64 each) / 4 waves per block
  // Q projection folds in the 1/sqrt(Dh)=0.125 attention scale.
  mha_gemm_bf16<0><<<256, 128, 0, stream>>>(Xq, Wqb, bq, Qh, 0.125f);
  mha_gemm_bf16<0><<<256, 128, 0, stream>>>(Xk, Wkb, bk, Kh, 1.0f);
  mha_gemm_bf16<1><<<256, 128, 0, stream>>>(Xv, Wvb, bv, Vt, 1.0f);

  // 4096 query-tile waves / 4 per block
  mha_attn<<<1024, 128, 0, stream>>>(Qh, Kh, Vt, maskp, Ctx);

  mha_gemm_bf16<2><<<256, 128, 0, stream>>>(Ctx, Wob, bo, d_out, 1.0f);
}
